// CRF_39273180955619
// MI455X (gfx1250) — matrix operand order repaired and checked by reading, not compiled, not run on hardware
//
#include <hip/hip_runtime.h>
#include <hip/hip_bf16.h>

// ---- problem constants (from reference) ----
#define BB 256    // batch
#define MM 1024   // seq len
#define DD 128    // feature dim
#define NN 26     // num labels
#define NP 32     // padded label dim

typedef __attribute__((ext_vector_type(16))) _Float16 v16h;
typedef __attribute__((ext_vector_type(8)))  float    v8f;

// --------------------------------------------------------------------------
// Kernel 0: zero the scalar output (atomicAdd target).
// --------------------------------------------------------------------------
__global__ void crf_zero_out(float* out) {
    if (threadIdx.x == 0 && blockIdx.x == 0) out[0] = 0.0f;
}

// --------------------------------------------------------------------------
// Kernel 1: emission GEMM  E[B*M, 32] = X[B*M, 128] @ W^T  (cols >= 26 junk)
// 128 threads = 4 waves; block computes a 32(row) x 32(col) tile.
// wave (w>>1) picks row half, (w&1) picks col half -> each wave one 16x16 tile.
// f16 inputs staged in LDS, f32 WMMA accumulation over K=128 in 4 chunks.
// --------------------------------------------------------------------------
__global__ __launch_bounds__(128)
void crf_emission_gemm(const float* __restrict__ Xg,
                       const float* __restrict__ Wg,
                       float* __restrict__ E) {
    __shared__ _Float16 sA[32 * DD];   // 32 rows of X, f16
    __shared__ _Float16 sW[32 * DD];   // 26 rows of W + 6 zero pad rows

    const int tid = threadIdx.x;
    const int R   = blockIdx.x * 32;   // base row in flattened [B*M]

    // cooperative stage: X rows -> sA, W (zero-padded) -> sW
    for (int idx = tid; idx < 32 * DD; idx += 128) {
        const int r = idx >> 7, k = idx & (DD - 1);
        sA[idx] = (_Float16)Xg[(size_t)(R + r) * DD + k];
        sW[idx] = (_Float16)((r < NN) ? Wg[r * DD + k] : 0.0f);
    }
    __syncthreads();

    const int wave    = tid >> 5;
    const int lane    = tid & 31;
    const int lo16    = lane & 15;
    const bool hiHalf = lane >= 16;
    const int rowTile = (wave >> 1) * 16;   // 0 or 16
    const int colTile = (wave & 1) * 16;    // 0 or 16

    v8f acc = {};
    #pragma unroll
    for (int kc = 0; kc < DD; kc += 32) {
        // A fragment: 16x32 f16, M = rowTile+lo16
        // lanes 0-15: K = kc+0..15; lanes 16-31: K = kc+16..31
        const int m   = rowTile + lo16;
        const int kbA = kc + (hiHalf ? 16 : 0);
        const _Float16* pa = &sA[m * DD + kbA];
        v16h a;
        #pragma unroll
        for (int e = 0; e < 16; ++e) a[e] = pa[e];

        // B fragment: 32x16 f16, B[k][n] = W[n][k]; N = colTile+lo16
        // lanes 0-15 hold K = kc+0..15; lanes 16-31 hold K = kc+16..31
        const int n   = colTile + lo16;
        const int kbB = kc + (hiHalf ? 16 : 0);
        const _Float16* pb = &sW[n * DD + kbB];
        v16h b;
        #pragma unroll
        for (int e = 0; e < 16; ++e) b[e] = pb[e];

        acc = __builtin_amdgcn_wmma_f32_16x16x32_f16(
            /*neg_a=*/false, a, /*neg_b=*/false, b,
            /*c_mod=*/(short)0, acc, /*reuse_a=*/false, /*reuse_b=*/false);
    }

    // C layout: VGPR r -> M = r (lanes 0-15) / 8+r (lanes 16-31), N = lane&15
    const int colOut = colTile + lo16;
    #pragma unroll
    for (int r = 0; r < 8; ++r) {
        const int rowOut = R + rowTile + r + (hiHalf ? 8 : 0);
        E[(size_t)rowOut * NP + colOut] = acc[r];
    }
}

// --------------------------------------------------------------------------
// Kernel 2: per-sequence CRF forward recurrence (log space).
// One wave32 block per sequence; lane a owns label a (a < 26 active).
// lf shared across lanes via LDS broadcast each step.
// --------------------------------------------------------------------------
__global__ __launch_bounds__(32)
void crf_forward_seq(const float* __restrict__ E,
                     const int*   __restrict__ labels,
                     const float* __restrict__ Tg,
                     float* __restrict__ out) {
    const int seq = blockIdx.x;
    const int a   = threadIdx.x;
    const bool act = (a < NN);
    const float NEG = -1e30f;

    // lane a holds column a of T: Tcol[b] = T[b][a]
    float Tcol[NN];
    #pragma unroll
    for (int b = 0; b < NN; ++b) Tcol[b] = act ? Tg[b * NN + a] : 0.0f;

    __shared__ float lfs[32];

    const float* Eseq = E + (size_t)seq * MM * NP;
    const int*   Yseq = labels + (size_t)seq * MM;

    float lf = act ? Eseq[a] : NEG;    // lf[a] = e[0][a]
    float unnorm = 0.0f;
    int yprev = Yseq[0];
    if (a == yprev) unnorm += lf;      // e[0][y0]

    for (int i = 1; i < MM; ++i) {
        lfs[a] = lf;
        __syncthreads();

        // prefetch a future emission row (coalesced 128B rows)
        if (i + 8 < MM) __builtin_prefetch(&Eseq[(i + 8) * NP + a], 0, 0);

        const float e_i = Eseq[i * NP + a];   // lanes>=26 read finite pad (0)

        float mx = NEG;
        #pragma unroll
        for (int b = 0; b < NN; ++b) mx = fmaxf(mx, lfs[b] + Tcol[b]);
        float s = 0.0f;
        #pragma unroll
        for (int b = 0; b < NN; ++b) s += __expf(lfs[b] + Tcol[b] - mx);
        const float lfn = e_i + mx + __logf(s);

        const int y = Yseq[i];
        if (a == y) unnorm += e_i + Tcol[yprev];
        yprev = y;
        lf = act ? lfn : NEG;
        __syncthreads();
    }

    // logZ = logsumexp over lf (wave32 shuffle reduction; inactive lanes = NEG)
    float mx = lf;
    #pragma unroll
    for (int o = 16; o > 0; o >>= 1) mx = fmaxf(mx, __shfl_xor(mx, o, 32));
    float s = act ? __expf(lf - mx) : 0.0f;
    #pragma unroll
    for (int o = 16; o > 0; o >>= 1) s += __shfl_xor(s, o, 32);
    const float logZ = mx + __logf(s);

    float un = unnorm;
    #pragma unroll
    for (int o = 16; o > 0; o >>= 1) un += __shfl_xor(un, o, 32);

    if (a == 0) atomicAdd(out, (un - logZ) * (1.0f / (float)BB));
}

// --------------------------------------------------------------------------
extern "C" void kernel_launch(void* const* d_in, const int* in_sizes, int n_in,
                              void* d_out, int out_size, void* d_ws, size_t ws_size,
                              hipStream_t stream) {
    (void)in_sizes; (void)n_in; (void)out_size; (void)ws_size;
    const float* X      = (const float*)d_in[0];  // [B, M, D] f32
    const int*   labels = (const int*)  d_in[1];  // [B, M] i32
    const float* W      = (const float*)d_in[2];  // [N, D] f32
    const float* T      = (const float*)d_in[3];  // [N, N] f32
    float*       out    = (float*)d_out;          // scalar
    float*       E      = (float*)d_ws;           // [B*M, 32] f32 (32 MB)

    crf_zero_out<<<1, 32, 0, stream>>>(out);

    const int rowsTotal = BB * MM;                // 262144
    crf_emission_gemm<<<rowsTotal / 32, 128, 0, stream>>>(X, W, E);

    crf_forward_seq<<<BB, 32, 0, stream>>>(E, labels, T, out);
}
